// W_MSA_21225728377065
// MI455X (gfx1250) — matrix operand order repaired and checked by reading, not compiled, not run on hardware
//
#include <hip/hip_runtime.h>

typedef __attribute__((ext_vector_type(16))) _Float16 v16h;
typedef __attribute__((ext_vector_type(8)))  _Float16 v8h;
typedef __attribute__((ext_vector_type(8)))  float    v8f;
typedef __attribute__((ext_vector_type(4)))  unsigned int v4u;
typedef __attribute__((ext_vector_type(8)))  int      v8i;
typedef __attribute__((ext_vector_type(4)))  int      v4i;

#define NTOK 49
#define NPAD 64
#define C_DIM 384
#define NH 12
#define HD 32

// LDS strides in halves (padded so row->bank mapping is conflict-free)
#define SX_STRIDE  392   // 196 dwords, 196%64=4
#define SQK_STRIDE 40    // 20 dwords
#define SVT_STRIDE 72    // 36 dwords
#define SP_STRIDE  72

// LDS layout (offsets in halves), X buffer is reused as the O buffer.
// The fp32 TDM staging region (49x384 f32 = 75264B) aliases sQ+sK (122880B),
// which are only written after staging has been consumed.
#define OFF_X   0
#define OFF_Q   (OFF_X  + NPAD * SX_STRIDE)            // 25088
#define OFF_K   (OFF_Q  + NH * NPAD * SQK_STRIDE)      // +30720
#define OFF_VT  (OFF_K  + NH * NPAD * SQK_STRIDE)      // +30720
#define OFF_P   (OFF_VT + NH * HD * SVT_STRIDE)        // +27648
#define LDS_HALVES (OFF_P + 8 * 16 * SP_STRIDE)        // 123392 halves = 246784 B

#define STAGE_OFF_BYTES (OFF_Q * 2)                    // 50176

#define QKV_PACKED_HALVES (384 * 1152)   // 442368
#define PROJ_PACKED_HALVES (384 * 384)   // 147456

#if __has_builtin(__builtin_amdgcn_tensor_load_to_lds) && \
    __has_builtin(__builtin_amdgcn_s_wait_tensorcnt)
#define HAVE_TDM 1
#else
#define HAVE_TDM 0
#endif

// ---- WMMA fragment loaders (ISA 7.12.2, 16-bit, wave32) -------------------
// A (16x32): lane holds row = l&15; elem e: K = 16*half + e  (contiguous 16)
__device__ __forceinline__ v16h lds_a16(const _Float16* base, int stride,
                                        int row, int k0, int halfid) {
  const _Float16* p = base + row * stride + k0 + halfid * 16;
  v8h lo = *(const v8h*)p;
  v8h hi = *(const v8h*)(p + 8);
  return __builtin_shufflevector(lo, hi, 0,1,2,3,4,5,6,7,8,9,10,11,12,13,14,15);
}
// B (32x16): lane holds col = l&15; elem e: K = 16*half + e  (contiguous 16)
__device__ __forceinline__ v16h lds_b16(const _Float16* base, int stride,
                                        int col, int k0, int halfid) {
  const _Float16* p = base + col * stride + k0 + halfid * 16;
  v8h lo = *(const v8h*)p;
  v8h hi = *(const v8h*)(p + 8);
  return __builtin_shufflevector(lo, hi, 0,1,2,3,4,5,6,7,8,9,10,11,12,13,14,15);
}

__device__ __forceinline__ v8f wmma_f16(v16h a, v16h b, v8f c) {
  return __builtin_amdgcn_wmma_f32_16x16x32_f16(false, a, false, b, (short)0, c,
                                                false, false);
}

// ---- Weight pre-pack: fp32 row-major [K][N] -> f16 B-fragment tiles --------
// tile = nt*12 + kt (K-tiles contiguous for streaming), 512 halves per tile,
// within-tile order: lane-major then e  =>  one 32B contiguous load per lane.
__global__ void wmsa_pack_w(const float* __restrict__ w, _Float16* __restrict__ dst,
                            int ncols, int total) {
  int gid = blockIdx.x * 256 + threadIdx.x;
  if (gid >= total) return;
  int tile   = gid >> 9;
  int within = gid & 511;
  int lane   = within >> 4;
  int e      = within & 15;
  int nt = tile / 12, kt = tile - nt * 12;
  int K = 32 * kt + 16 * (lane >> 4) + e;
  int N = 16 * nt + (lane & 15);
  dst[gid] = (_Float16)w[(size_t)K * ncols + N];
}

// ---- Fused W-MSA: one workgroup (8 wave32) per 7x7 window ------------------
__global__ __launch_bounds__(256, 1)
void wmsa_fused(const float* __restrict__ x,
                const float* __restrict__ qkv_b,
                const float* __restrict__ proj_b,
                const _Float16* __restrict__ wq,   // packed qkv weights
                const _Float16* __restrict__ wp,   // packed proj weights
                float* __restrict__ out) {
  extern __shared__ _Float16 smem[];
  _Float16* sX  = smem + OFF_X;    // 64 x 384 (stride 392); reused as O buffer
  _Float16* sQ  = smem + OFF_Q;    // [12][64][32] stride 40
  _Float16* sK  = smem + OFF_K;    // [12][64][32] stride 40
  _Float16* sVt = smem + OFF_VT;   // [12][32][64] stride 72 (V transposed)
  _Float16* sP  = smem + OFF_P;    // per-wave 16x64 stride 72

  const int tid    = threadIdx.x;
  const int lane16 = tid & 15;
  const int halfid = (tid >> 4) & 1;
  const int wave   = tid >> 5;
  const int w0 = blockIdx.x;            // 0..2047
  const int bb = w0 >> 6;               // batch
  const int wy = (w0 >> 3) & 7;         // window y
  const int wx = w0 & 7;                // window x

#if HAVE_TDM
  // ---- stage 0 (TDM): DMA the 7x7 window (7 lines x 2688 f32, stride 21504)
  // straight into LDS staging, then convert f32 -> padded f16 tile.
  if (wave == 0) {
    unsigned long long ga = (unsigned long long)(uintptr_t)x +
        4ull * (((unsigned long long)bb * 3136 + (wy * 7) * 56 + wx * 7) * C_DIM);
    v4u g0;
    g0[0] = 1u;                                    // count=1, user descriptor
    g0[1] = (unsigned)STAGE_OFF_BYTES;             // lds_addr (bytes)
    g0[2] = (unsigned)(ga & 0xffffffffull);        // global_addr[31:0]
    g0[3] = (unsigned)((ga >> 32) & 0x1ffffffull)  // global_addr[56:32]
          | (2u << 30);                            // type=2 ("image")
    v8i g1;
    g1[0] = (int)(2u << 16);                       // data_size=2 -> 4B elems
    g1[1] = (int)(2688u << 16);                    // tensor_dim0[15:0]
    g1[2] = (int)(7u << 16);                       // dim0 hi=0 | tensor_dim1=7
    g1[3] = (int)(2688u << 16);                    // tile_dim0=2688
    g1[4] = 7;                                     // tile_dim1=7, tile_dim2=0
    g1[5] = 21504;                                 // tensor_dim0_stride lo
    g1[6] = 0;                                     // stride hi | dim1_stride lo
    g1[7] = 0;
    v4i gz = (v4i){};
#if __clang_major__ >= 23
    v8i gz8 = (v8i){};
    __builtin_amdgcn_tensor_load_to_lds(g0, g1, gz, gz, gz8, 0);
#else
    __builtin_amdgcn_tensor_load_to_lds(g0, g1, gz, gz, 0);
#endif
    __builtin_amdgcn_s_wait_tensorcnt(0);
  } else {
    // warm L2 with this wave's first packed-weight tile while the DMA flies
    __builtin_prefetch(wq + ((wave * 9 * 12) << 9) + ((tid & 31) << 4), 0, 3);
  }
  __syncthreads();
  {
    const float* stage = (const float*)(smem + OFF_Q);
    for (int idx = tid; idx < NPAD * C_DIM; idx += 256) {
      int n = idx / C_DIM, c = idx - n * C_DIM;
      float v = (n < NTOK) ? stage[n * C_DIM + c] : 0.f;
      sX[n * SX_STRIDE + c] = (_Float16)v;
    }
  }
  __syncthreads();
#else
  // ---- stage 0 (fallback): window-partition load via VGPRs -----------------
  for (int idx = tid; idx < NPAD * C_DIM; idx += 256) {
    int n = idx / C_DIM, c = idx - n * C_DIM;
    float v = 0.f;
    if (n < NTOK) {
      int iy = n / 7, ix = n - iy * 7;
      int L = (wy * 7 + iy) * 56 + wx * 7 + ix;
      v = x[((size_t)bb * 3136 + L) * C_DIM + c];
    }
    sX[n * SX_STRIDE + c] = (_Float16)v;
  }
  __syncthreads();
#endif

  // ---- stage 1: QKV = X @ Wqkv (+b); Q scaled; V stored transposed ---------
  const float scale = 0.17677669529663687f;  // 32^-0.5
  for (int ci = 0; ci < 9; ++ci) {           // 72 col-tiles over 8 waves
    int ct = wave * 9 + ci;
    v8f acc[4];
    #pragma unroll
    for (int rt = 0; rt < 4; ++rt) acc[rt] = (v8f){};
    for (int k = 0; k < 12; ++k) {
      v16h b = *(const v16h*)(wq + (((ct * 12 + k) << 9) + ((tid & 31) << 4)));
      #pragma unroll
      for (int rt = 0; rt < 4; ++rt) {
        v16h a = lds_a16(sX, SX_STRIDE, 16 * rt + lane16, 32 * k, halfid);
        acc[rt] = wmma_f16(a, b, acc[rt]);
      }
    }
    int s = ct / 24;                 // 0=q,1=k,2=v
    int h = (ct % 24) >> 1;          // head
    int d = ((ct & 1) << 4) + lane16;
    float bias = qkv_b[ct * 16 + lane16];
    float mul  = (s == 0) ? scale : 1.f;
    #pragma unroll
    for (int rt = 0; rt < 4; ++rt) {
      if (s == 2) {   // V transposed: 8 consecutive n per lane -> one 16B store
        v8h tmp;
        #pragma unroll
        for (int r = 0; r < 8; ++r) tmp[r] = (_Float16)(acc[rt][r] + bias);
        *(v8h*)&sVt[(h * HD + d) * SVT_STRIDE + 16 * rt + 8 * halfid] = tmp;
      } else {
        _Float16* dst = (s == 0 ? sQ : sK) + h * NPAD * SQK_STRIDE + d;
        #pragma unroll
        for (int r = 0; r < 8; ++r) {
          int n = 16 * rt + 8 * halfid + r;
          dst[n * SQK_STRIDE] = (_Float16)((acc[rt][r] + bias) * mul);
        }
      }
    }
  }
  __syncthreads();

  // ---- stage 2: per-head attention (S=QK^T, softmax, O=PV) -----------------
  _Float16* pbuf = sP + wave * 16 * SP_STRIDE;
  for (int h = wave; h < NH; h += 8) {
    const _Float16* qh = sQ + h * NPAD * SQK_STRIDE;
    const _Float16* kh = sK + h * NPAD * SQK_STRIDE;
    const _Float16* vh = sVt + h * HD * SVT_STRIDE;
    for (int rt = 0; rt < 4; ++rt) {
      v16h qf = lds_a16(qh, SQK_STRIDE, 16 * rt + lane16, 0, halfid);
      v8f s4[4];
      #pragma unroll
      for (int ct = 0; ct < 4; ++ct) {
        v16h kf = lds_b16(kh, SQK_STRIDE, 16 * ct + lane16, 0, halfid);
        v8f z = (v8f){};
        s4[ct] = wmma_f16(qf, kf, z);
      }
      // mask padded key columns (n >= 49)
      #pragma unroll
      for (int ct = 0; ct < 4; ++ct)
        if (16 * ct + lane16 >= NTOK) {
          #pragma unroll
          for (int r = 0; r < 8; ++r) s4[ct][r] = -1e30f;
        }
      // row softmax: rows live across 16 lanes of one half + 4 col-tiles
      #pragma unroll
      for (int r = 0; r < 8; ++r) {
        float m = fmaxf(fmaxf(s4[0][r], s4[1][r]), fmaxf(s4[2][r], s4[3][r]));
        for (int o = 8; o >= 1; o >>= 1) m = fmaxf(m, __shfl_xor(m, o, 16));
        float e[4], sum = 0.f;
        #pragma unroll
        for (int ct = 0; ct < 4; ++ct) { e[ct] = __expf(s4[ct][r] - m); sum += e[ct]; }
        for (int o = 8; o >= 1; o >>= 1) sum += __shfl_xor(sum, o, 16);
        float inv = 1.f / sum;
        #pragma unroll
        for (int ct = 0; ct < 4; ++ct) s4[ct][r] = e[ct] * inv;
      }
      // spill P (16x64 f16) to per-wave scratch for re-fragmenting as A
      #pragma unroll
      for (int ct = 0; ct < 4; ++ct)
        #pragma unroll
        for (int r = 0; r < 8; ++r)
          pbuf[(8 * halfid + r) * SP_STRIDE + 16 * ct + lane16] = (_Float16)s4[ct][r];
      // O(16x32) = P(16x64) @ V(64x32)
      v16h pf0 = lds_a16(pbuf, SP_STRIDE, lane16, 0, halfid);
      v16h pf1 = lds_a16(pbuf, SP_STRIDE, lane16, 32, halfid);
      #pragma unroll
      for (int cv = 0; cv < 2; ++cv) {
        v16h vf0 = lds_b16(vh, SVT_STRIDE, 16 * cv + lane16, 0, halfid);
        v16h vf1 = lds_b16(vh, SVT_STRIDE, 16 * cv + lane16, 32, halfid);
        v8f z = (v8f){};
        v8f o = wmma_f16(pf0, vf0, z);
        o = wmma_f16(pf1, vf1, o);
        int col = h * HD + 16 * cv + lane16;
        #pragma unroll
        for (int r = 0; r < 8; ++r)
          sX[(16 * rt + 8 * halfid + r) * SX_STRIDE + col] = (_Float16)o[r];
      }
    }
  }
  __syncthreads();

  // ---- stage 3: out = O @ Wproj + b, window-reverse scatter ----------------
  for (int t = wave; t < 96; t += 8) {   // 4 row-tiles x 24 col-tiles
    int rt = t & 3, ct = t >> 2;
    v8f acc = (v8f){};
    for (int k = 0; k < 12; ++k) {
      v16h a = lds_a16(sX, SX_STRIDE, 16 * rt + lane16, 32 * k, halfid);
      v16h b = *(const v16h*)(wp + (((ct * 12 + k) << 9) + ((tid & 31) << 4)));
      acc = wmma_f16(a, b, acc);
    }
    int col = ct * 16 + lane16;
    float bias = proj_b[col];
    #pragma unroll
    for (int r = 0; r < 8; ++r) {
      int n = 16 * rt + 8 * halfid + r;
      if (n < NTOK) {
        int iy = n / 7, ix = n - iy * 7;
        int L = (wy * 7 + iy) * 56 + wx * 7 + ix;
        out[((size_t)bb * 3136 + L) * C_DIM + col] = acc[r] + bias;
      }
    }
  }
}

extern "C" void kernel_launch(void* const* d_in, const int* in_sizes, int n_in,
                              void* d_out, int out_size, void* d_ws, size_t ws_size,
                              hipStream_t stream) {
  (void)in_sizes; (void)n_in; (void)out_size; (void)ws_size;
  const float* x      = (const float*)d_in[0];
  const float* qkv_w  = (const float*)d_in[1];
  const float* qkv_b  = (const float*)d_in[2];
  const float* proj_w = (const float*)d_in[3];
  const float* proj_b = (const float*)d_in[4];

  _Float16* wq = (_Float16*)d_ws;                  // 884736 B
  _Float16* wp = wq + QKV_PACKED_HALVES;           // 294912 B  (total ~1.2 MB)

  wmsa_pack_w<<<(QKV_PACKED_HALVES + 255) / 256, 256, 0, stream>>>(
      qkv_w, wq, 1152, QKV_PACKED_HALVES);
  wmsa_pack_w<<<(PROJ_PACKED_HALVES + 255) / 256, 256, 0, stream>>>(
      proj_w, wp, 384, PROJ_PACKED_HALVES);

  size_t shmem = (size_t)LDS_HALVES * sizeof(_Float16);  // 246784 B (<320KB/WGP)
  hipFuncSetAttribute((const void*)wmsa_fused,
                      hipFuncAttributeMaxDynamicSharedMemorySize, (int)shmem);
  wmsa_fused<<<2048, 256, shmem, stream>>>(x, qkv_b, proj_b, wq, wp, (float*)d_out);
}
